// GeoAwarePooling_45792941310700
// MI455X (gfx1250) — matrix operand order repaired and checked by reading, not compiled, not run on hardware
//
#include <hip/hip_runtime.h>
#include <hip/hip_bf16.h>
#include <cfloat>
#include <cmath>

// ---------------- problem constants (from reference) ----------------
#define BB   8
#define VV   4
#define NN   16384
#define CP   256
#define TILE 32          // points per block
#define LDSTR 260        // f32 working buffer row stride (floats)
#define HSTR  264        // f16 A-operand buffer row stride (halves, mult of 8 -> 16B aligned)
#define LN_EPS 1e-5f

typedef float     v8f  __attribute__((ext_vector_type(8)));
typedef _Float16  v8h  __attribute__((ext_vector_type(8)));
typedef _Float16  v16h __attribute__((ext_vector_type(16)));

// monotone f32 <-> u32 encoding so atomicMax(u32) == float max (handles negatives)
__device__ __forceinline__ unsigned encF(float f) {
    unsigned u = __float_as_uint(f);
    return (u & 0x80000000u) ? ~u : (u | 0x80000000u);
}
__device__ __forceinline__ float decF(unsigned u) {
    unsigned v = (u & 0x80000000u) ? (u & 0x7FFFFFFFu) : ~u;
    return __uint_as_float(v);
}
#define ENC_NEG_INF 0x007FFFFFu   // encF(-inf)

// ---------------- block reductions for the stats kernel ----------------
__device__ __forceinline__ float brSum(float v, float* red) {
    int t = threadIdx.x; red[t] = v; __syncthreads();
    for (int s = 128; s > 0; s >>= 1) { if (t < s) red[t] += red[t + s]; __syncthreads(); }
    float r = red[0]; __syncthreads(); return r;
}
__device__ __forceinline__ float brMax(float v, float* red) {
    int t = threadIdx.x; red[t] = v; __syncthreads();
    for (int s = 128; s > 0; s >>= 1) { if (t < s) red[t] = fmaxf(red[t], red[t + s]); __syncthreads(); }
    float r = red[0]; __syncthreads(); return r;
}
__device__ __forceinline__ float brMin(float v, float* red) {
    int t = threadIdx.x; red[t] = v; __syncthreads();
    for (int s = 128; s > 0; s >>= 1) { if (t < s) red[t] = fminf(red[t], red[t + s]); __syncthreads(); }
    float r = red[0]; __syncthreads(); return r;
}

// ---------------- workspace init ----------------
__global__ void gap_init_kernel(unsigned* gxEnc, float* pooled, float* denom) {
    int i = blockIdx.x * 256 + threadIdx.x;      // 8192 threads
    gxEnc[i]  = ENC_NEG_INF;
    pooled[i] = 0.0f;
    if (i < BB * VV) denom[i] = 0.0f;
}

// ---------------- repack f32 weight [256x256] into f16 B-fragment layout ------
// Wt[kb][c][k] : for k-block kb (32 K's), column c, the 32 values W[kb*32+k][c]
// contiguous (64B). B frag per lane = 32B at ((kb*256+c)*32 + hi*16)*2 bytes.
__global__ __launch_bounds__(256) void gap_prepw_kernel(const float* __restrict__ W,
                                                        _Float16* __restrict__ Wt) {
    int kb = blockIdx.x;            // 0..7
    int c  = threadIdx.x;           // 0..255
    _Float16* dst = Wt + ((size_t)kb * 256 + c) * 32;
#pragma unroll 8
    for (int k = 0; k < 32; ++k)
        dst[k] = (_Float16)W[(kb * 32 + k) * 256 + c];
}

// ---------------- per-(b,v) center + diameter ----------------
__global__ __launch_bounds__(256) void gap_stats_kernel(
        const float* __restrict__ xyz, const float* __restrict__ masks,
        float* __restrict__ centerDiam) {
    int bv = blockIdx.x;            // 0..31
    int b  = bv >> 2;
    int t  = threadIdx.x;
    const float* mrow = masks + (size_t)bv * NN;
    const float* xb   = xyz + (size_t)b * 3 * NN;

    float cnt = 0.f, sx = 0.f, sy = 0.f, sz = 0.f;
    float mxx = -FLT_MAX, mxy = -FLT_MAX, mxz = -FLT_MAX;
    float mnx =  FLT_MAX, mny =  FLT_MAX, mnz =  FLT_MAX;
    for (int n = t; n < NN; n += 256) {
        float m = mrow[n];
        float x = xb[n], y = xb[NN + n], z = xb[2 * NN + n];
        cnt += m; sx += x * m; sy += y * m; sz += z * m;
        float xm = x * m, ym = y * m, zm = z * m;   // masked-out points contribute exactly 0
        mxx = fmaxf(mxx, xm); mxy = fmaxf(mxy, ym); mxz = fmaxf(mxz, zm);
        mnx = fminf(mnx, xm); mny = fminf(mny, ym); mnz = fminf(mnz, zm);
    }
    __shared__ float red[256];
    cnt = brSum(cnt, red); sx = brSum(sx, red); sy = brSum(sy, red); sz = brSum(sz, red);
    mxx = brMax(mxx, red); mxy = brMax(mxy, red); mxz = brMax(mxz, red);
    mnx = brMin(mnx, red); mny = brMin(mny, red); mnz = brMin(mnz, red);
    if (t == 0) {
        float valid = fmaxf(cnt, 1.0f);
        float d = fmaxf(fmaxf(mxx - mnx, mxy - mny), mxz - mnz);
        if (d == 0.0f) d = 1.0f;
        centerDiam[bv * 4 + 0] = sx / valid;
        centerDiam[bv * 4 + 1] = sy / valid;
        centerDiam[bv * 4 + 2] = sz / valid;
        centerDiam[bv * 4 + 3] = d;
    }
}

// ---------------- LN (+optional ReLU) over the 256 channels of each point ----
// thread (p = t>>3, q = t&7) owns hbuf[p][q*32 .. +31]; 8-lane shuffle stats.
// Writes the f32 result to hbuf and an f16 copy to habuf (WMMA A operand).
__device__ __forceinline__ void gap_ln_phase(float* hbuf, _Float16* habuf,
                                             const float* __restrict__ g,
                                             const float* __restrict__ be, bool doRelu) {
    int t = threadIdx.x, p = t >> 3, q = t & 7, c0 = q * 32;
    float* row = hbuf + p * LDSTR + c0;
    _Float16* hrow = habuf + p * HSTR + c0;
    float s1 = 0.f, s2 = 0.f;
#pragma unroll
    for (int i = 0; i < 32; ++i) { float y = row[i]; s1 += y; s2 += y * y; }
    s1 += __shfl_xor(s1, 1); s1 += __shfl_xor(s1, 2); s1 += __shfl_xor(s1, 4);
    s2 += __shfl_xor(s2, 1); s2 += __shfl_xor(s2, 2); s2 += __shfl_xor(s2, 4);
    float mean = s1 * (1.0f / CP);
    float var  = s2 * (1.0f / CP) - mean * mean;
    float rstd = rsqrtf(var + LN_EPS);
#pragma unroll
    for (int i = 0; i < 32; ++i) {
        float y = (row[i] - mean) * rstd * g[c0 + i] + be[c0 + i];
        if (doRelu) y = fmaxf(y, 0.0f);
        row[i]  = y;
        hrow[i] = (_Float16)y;
    }
    __syncthreads();
}

// ---------------- 32x256 = (32x256 f16 in LDS) @ (256x256 f16 Wt) + bias ------
// f16 WMMA 16x16x32, f32 accumulate. wave w: rows pg = w>>2 (16), cols ch = w&3 (64).
// A frag (16-bit 16x32): lanes 0-15 K=0-15, lanes 16-31 K=16-31, M = lane&15.
// B frag (32x16): lanes 0-15 K=0-15, lanes 16-31 K=16-31 -> 32B contiguous in Wt.
__device__ __forceinline__ void gap_gemm256(const _Float16* __restrict__ Wt,
                                            const float* __restrict__ bias,
                                            float* hbuf, const _Float16* habuf) {
    int t = threadIdx.x, lane = t & 31, lo = lane & 15, hi = lane >> 4;
    int w = t >> 5, pg = w >> 2, ch = w & 3;
    int rowA = pg * 16 + lo;
    const _Float16* arow = habuf + rowA * HSTR;
    v8f acc[4] = {};
#pragma unroll
    for (int kb = 0; kb < 8; ++kb) {
        int k0 = kb * 32;
        v8h a0 = *(const v8h*)(arow + k0 + hi * 16);
        v8h a1 = *(const v8h*)(arow + k0 + hi * 16 + 8);
        v16h a = __builtin_shufflevector(a0, a1, 0, 1, 2, 3, 4, 5, 6, 7,
                                                 8, 9, 10, 11, 12, 13, 14, 15);
#pragma unroll
        for (int t8 = 0; t8 < 4; ++t8) {
            int col = ch * 64 + t8 * 16 + lo;
            const _Float16* bp = Wt + ((size_t)(kb * 256 + col)) * 32 + hi * 16;
            v8h b0 = *(const v8h*)(bp);
            v8h b1 = *(const v8h*)(bp + 8);
            v16h bfrag = __builtin_shufflevector(b0, b1, 0, 1, 2, 3, 4, 5, 6, 7,
                                                         8, 9, 10, 11, 12, 13, 14, 15);
            acc[t8] = __builtin_amdgcn_wmma_f32_16x16x32_f16(
                false, a, false, bfrag, (short)0, acc[t8], false, false);
        }
    }
    __syncthreads();   // all waves done reading habuf
#pragma unroll
    for (int t8 = 0; t8 < 4; ++t8) {
        int col = ch * 64 + t8 * 16 + lo;
        float bcol = bias[col];
#pragma unroll
        for (int r = 0; r < 8; ++r) {        // f32 C layout: VGPR r -> M=r / r+8
            int row = pg * 16 + hi * 8 + r;
            hbuf[row * LDSTR + col] = acc[t8][r] + bcol;
        }
    }
    __syncthreads();
}

// ---------------- norm_xyz -> Linear -> LN -> ReLU -> Linear -> LN into hbuf ----
__device__ __forceinline__ void gap_compute_local(
        const float* __restrict__ xyz,
        const float* __restrict__ W1, const float* __restrict__ b1,
        const float* __restrict__ g1, const float* __restrict__ be1,
        const _Float16* __restrict__ Wt2, const float* __restrict__ b2,
        const float* __restrict__ g2, const float* __restrict__ be2,
        const float* __restrict__ centerDiam,
        float* hbuf, _Float16* habuf, int b, int bv, int n0) {
    int t = threadIdx.x, p = t >> 3, q = t & 7, c0 = q * 32;
    float cx = centerDiam[bv * 4 + 0], cy = centerDiam[bv * 4 + 1];
    float cz = centerDiam[bv * 4 + 2];
    float invd = 1.0f / centerDiam[bv * 4 + 3];
    int n = n0 + p;
    const float* xb = xyz + (size_t)b * 3 * NN;
    float nx = (xb[n] - cx) * invd;
    float ny = (xb[NN + n] - cy) * invd;
    float nz = (xb[2 * NN + n] - cz) * invd;
    float* row = hbuf + p * LDSTR + c0;
#pragma unroll 8
    for (int i = 0; i < 32; ++i) {
        int c = c0 + i;
        row[i] = fmaf(nx, W1[c], fmaf(ny, W1[256 + c], fmaf(nz, W1[512 + c], b1[c])));
    }
    gap_ln_phase(hbuf, habuf, g1, be1, true);    // h = relu(LN(.))  (f32 + f16 copy)
    gap_gemm256(Wt2, b2, hbuf, habuf);           // y2 = h @ W2 + b2 (f16 WMMA, f32 acc)
    gap_ln_phase(hbuf, habuf, g2, be2, false);   // local = LN(y2)
}

// ---------------- pass 1: local_xyz -> channel max into gxEnc ----------------
__global__ __launch_bounds__(256) void gap_pass1_kernel(
        const float* __restrict__ xyz,
        const float* __restrict__ W1, const float* __restrict__ b1,
        const float* __restrict__ g1, const float* __restrict__ be1,
        const _Float16* __restrict__ Wt2, const float* __restrict__ b2,
        const float* __restrict__ g2, const float* __restrict__ be2,
        const float* __restrict__ centerDiam, unsigned* __restrict__ gxEnc) {
    __shared__ float hbuf[TILE * LDSTR];
    __shared__ __align__(16) _Float16 habuf[TILE * HSTR];
    int id = blockIdx.x;
    int v = id & 3, b = (id >> 2) & 7, nt = id >> 5;
    int bv = b * 4 + v, n0 = nt * TILE;
    gap_compute_local(xyz, W1, b1, g1, be1, Wt2, b2, g2, be2, centerDiam,
                      hbuf, habuf, b, bv, n0);
    int t = threadIdx.x;                 // one channel per thread
    float m = -FLT_MAX;
#pragma unroll 8
    for (int p = 0; p < TILE; ++p) m = fmaxf(m, hbuf[p * LDSTR + t]);
    atomicMax(&gxEnc[bv * 256 + t], encF(m));
}

// ---------------- fold global_xyz @ W3[256:512] + b3 into per-(b,v) bias ------
__global__ __launch_bounds__(256) void gap_gxw3_kernel(
        const unsigned* __restrict__ gxEnc, const float* __restrict__ W3,
        const float* __restrict__ b3, float* __restrict__ gxb) {
    int bv = blockIdx.x, t = threadIdx.x;
    __shared__ float gxs[256];
    gxs[t] = decF(gxEnc[bv * 256 + t]);
    __syncthreads();
    float acc = b3[t];
    for (int k = 0; k < 256; ++k)
        acc = fmaf(gxs[k], W3[(256 + k) * 256 + t], acc);
    gxb[bv * 256 + t] = acc;
}

// ---------------- pass 2: recompute local, proj2, weights, pooling -----------
__global__ __launch_bounds__(256) void gap_pass2_kernel(
        const float* __restrict__ xyz, const float* __restrict__ feats,
        const float* __restrict__ masks,
        const float* __restrict__ W1, const float* __restrict__ b1,
        const float* __restrict__ g1, const float* __restrict__ be1,
        const _Float16* __restrict__ Wt2, const float* __restrict__ b2,
        const float* __restrict__ g2, const float* __restrict__ be2,
        const _Float16* __restrict__ Wt3, const float* __restrict__ g3,
        const float* __restrict__ be3, const float* __restrict__ W4,
        const float* __restrict__ centerDiam, const float* __restrict__ gxb,
        float* __restrict__ pooled, float* __restrict__ denom) {
    __shared__ float hbuf[TILE * LDSTR];
    __shared__ __align__(16) _Float16 habuf[TILE * HSTR];
    __shared__ float sW[TILE];
    int id = blockIdx.x;
    int v = id & 3, b = (id >> 2) & 7, nt = id >> 5;   // v innermost -> L2 reuse of feats
    int bv = b * 4 + v, n0 = nt * TILE;
    gap_compute_local(xyz, W1, b1, g1, be1, Wt2, b2, g2, be2, centerDiam,
                      hbuf, habuf, b, bv, n0);
    gap_gemm256(Wt3, gxb + bv * 256, hbuf, habuf);  // y3 = local@W3[:256] + (gx@W3[256:]+b3)
    gap_ln_phase(hbuf, habuf, g3, be3, true);       // h2 = relu(LN(y3))

    int t = threadIdx.x, p = t >> 3, q = t & 7, c0 = q * 32;
    const float* row = hbuf + p * LDSTR + c0;
    float pd = 0.f;
#pragma unroll
    for (int i = 0; i < 32; ++i) pd = fmaf(row[i], W4[c0 + i], pd);
    pd += __shfl_xor(pd, 1); pd += __shfl_xor(pd, 2); pd += __shfl_xor(pd, 4);
    if (q == 0) {
        float wgt = 2.0f / (1.0f + expf(-pd));              // 2*sigmoid
        sW[p] = masks[(size_t)bv * NN + n0 + p] * wgt;      // mask * weight
    }
    __syncthreads();

    // pooling: thread t = channel c; 32 contiguous points -> float4 reads
    int c = t;
    const float* fp = feats + ((size_t)b * CP + c) * NN + n0;
    float acc = 0.f;
#pragma unroll
    for (int p8 = 0; p8 < TILE; p8 += 4) {
        float4 f = *(const float4*)(fp + p8);
        acc = fmaf(f.x, sW[p8 + 0], acc);
        acc = fmaf(f.y, sW[p8 + 1], acc);
        acc = fmaf(f.z, sW[p8 + 2], acc);
        acc = fmaf(f.w, sW[p8 + 3], acc);
    }
    atomicAdd(&pooled[bv * 256 + c], acc);
    if (t == 0) {
        float s = 0.f;
        for (int p2 = 0; p2 < TILE; ++p2) s += sW[p2];
        atomicAdd(&denom[bv], s);
    }
}

// ---------------- finalize ----------------
__global__ void gap_final_kernel(const float* __restrict__ pooled,
                                 const float* __restrict__ denom,
                                 const unsigned* __restrict__ gxEnc,
                                 float* __restrict__ out) {
    int i = blockIdx.x * 256 + threadIdx.x;   // 8192
    int bv = i >> 8;
    out[i] = pooled[i] / fmaxf(denom[bv], 1e-8f) + decF(gxEnc[i]);
}

extern "C" void kernel_launch(void* const* d_in, const int* in_sizes, int n_in,
                              void* d_out, int out_size, void* d_ws, size_t ws_size,
                              hipStream_t stream) {
    const float* xyz   = (const float*)d_in[0];
    const float* feats = (const float*)d_in[1];
    const float* masks = (const float*)d_in[2];
    const float* W1 = (const float*)d_in[3];
    const float* b1 = (const float*)d_in[4];
    const float* g1 = (const float*)d_in[5];
    const float* be1 = (const float*)d_in[6];
    const float* W2 = (const float*)d_in[7];
    const float* b2 = (const float*)d_in[8];
    const float* g2 = (const float*)d_in[9];
    const float* be2 = (const float*)d_in[10];
    const float* W3 = (const float*)d_in[11];
    const float* b3 = (const float*)d_in[12];
    const float* g3 = (const float*)d_in[13];
    const float* be3 = (const float*)d_in[14];
    const float* W4 = (const float*)d_in[15];
    float* out = (float*)d_out;

    // workspace layout (f32 units):
    // centerDiam[128] | gxEnc[8192] | pooled[8192] | denom[32] | gxb[8192] |
    // pad to 24768 | Wt2 (65536 f16 = 32768 f32) | Wt3 (65536 f16)
    float*    ws         = (float*)d_ws;
    float*    centerDiam = ws;
    unsigned* gxEnc      = (unsigned*)(ws + 128);
    float*    pooled     = ws + 128 + 8192;
    float*    denom      = pooled + 8192;
    float*    gxb        = denom + 32;
    _Float16* Wt2        = (_Float16*)(ws + 24768);       // 16B aligned
    _Float16* Wt3        = Wt2 + 65536;

    const int nBlocks = (NN / TILE) * BB * VV;   // 16384

    gap_init_kernel<<<32, 256, 0, stream>>>(gxEnc, pooled, denom);
    gap_prepw_kernel<<<8, 256, 0, stream>>>(W2, Wt2);
    gap_prepw_kernel<<<8, 256, 0, stream>>>(W3, Wt3);     // rows 0..255 (local half)
    gap_stats_kernel<<<BB * VV, 256, 0, stream>>>(xyz, masks, centerDiam);
    gap_pass1_kernel<<<nBlocks, 256, 0, stream>>>(
        xyz, W1, b1, g1, be1, Wt2, b2, g2, be2, centerDiam, gxEnc);
    gap_gxw3_kernel<<<BB * VV, 256, 0, stream>>>(gxEnc, W3, b3, gxb);
    gap_pass2_kernel<<<nBlocks, 256, 0, stream>>>(
        xyz, feats, masks, W1, b1, g1, be1, Wt2, b2, g2, be2,
        Wt3, g3, be3, W4, centerDiam, gxb, pooled, denom);
    gap_final_kernel<<<32, 256, 0, stream>>>(pooled, denom, gxEnc, out);
}
